// TemporalRetNet_40278203302378
// MI455X (gfx1250) — matrix operand order repaired and checked by reading, not compiled, not run on hardware
//
#include <hip/hip_runtime.h>
#include <math.h>

// ---------------------------------------------------------------------------
// CDNA5 (gfx1250) RetNet block: bf16 WMMA GEMMs + flash-retention.
// wave32; WMMA 16x16x32 bf16 -> f32 accum; async global->LDS staging when
// the toolchain exposes the gfx1250 builtins (guarded, sync fallback kept).
// ---------------------------------------------------------------------------

typedef __attribute__((ext_vector_type(16))) __bf16 v16bf;
typedef __attribute__((ext_vector_type(8)))  float  v8f;
typedef __attribute__((ext_vector_type(4)))  int    v4i;

#define D_MODEL   1024
#define NHEAD     16
#define HEAD_DIM  64
#define SEQ       2048
#define BATCH     2
#define ROWS      (BATCH * SEQ)      // 4096
#define HID       2730
#define HID_PAD   2752               // 86 * 32, zero-padded K for final GEMM
#define N_IN      5460
#define N_IN_PAD  5504               // 43 * 128, zero-padded N for w_in GEMM

#if defined(__has_builtin)
#if defined(__gfx1250__) && __has_builtin(__builtin_amdgcn_global_load_async_to_lds_b128) && \
    __has_builtin(__builtin_amdgcn_s_wait_asynccnt)
#define HAVE_ASYNC_LDS 1
#endif
#endif
#ifndef HAVE_ASYNC_LDS
#define HAVE_ASYNC_LDS 0
#endif

union AF { v16bf v; unsigned u[8]; };

static __device__ __forceinline__ v8f vzero8() {
    v8f z = {0.f, 0.f, 0.f, 0.f, 0.f, 0.f, 0.f, 0.f};
    return z;
}
static __device__ __forceinline__ float sigmoidf_(float x) {
    return 1.0f / (1.0f + __expf(-x));
}

#if HAVE_ASYNC_LDS
static __device__ __forceinline__ void async_copy_b128(void* lds_dst, const void* g_src) {
    // ROCm 7.2 signature: (v4i* global_src, v4i* lds_dst, imm offset, imm cpol)
    __builtin_amdgcn_global_load_async_to_lds_b128((v4i*)g_src, (v4i*)lds_dst, 0, 0);
}
static __device__ __forceinline__ void async_wait0() {
    __builtin_amdgcn_s_wait_asynccnt(0);
}
#endif

// ---------------------------------------------------------------------------
// f32 -> bf16 conversion with zero padding (padded weight buffers).
// ---------------------------------------------------------------------------
__global__ void cvt_pad_bf16(const float* __restrict__ src, __bf16* __restrict__ dst,
                             int srcR, int srcC, int dstC, long total) {
    long e = (long)blockIdx.x * blockDim.x + threadIdx.x;
    if (e >= total) return;
    int r = (int)(e / dstC), c = (int)(e % dstC);
    float v = (r < srcR && c < srcC) ? src[(long)r * srcC + c] : 0.0f;
    dst[e] = (__bf16)v;
}

// ---------------------------------------------------------------------------
// Row LayerNorm (D=1024), bf16 output.
// ---------------------------------------------------------------------------
__global__ __launch_bounds__(256) void layernorm_bf16_kernel(
    const float* __restrict__ x, const float* __restrict__ w,
    const float* __restrict__ b, __bf16* __restrict__ out) {
    __shared__ float s1[256], s2[256];
    int row = blockIdx.x, tid = threadIdx.x;
    const float* xr = x + (size_t)row * D_MODEL;
    float a = 0.f, q = 0.f;
    for (int i = tid; i < D_MODEL; i += 256) { float v = xr[i]; a += v; q += v * v; }
    s1[tid] = a; s2[tid] = q;
    __syncthreads();
    for (int off = 128; off; off >>= 1) {
        if (tid < off) { s1[tid] += s1[tid + off]; s2[tid] += s2[tid + off]; }
        __syncthreads();
    }
    float mu  = s1[0] * (1.0f / D_MODEL);
    float var = s2[0] * (1.0f / D_MODEL) - mu * mu;
    float rs  = rsqrtf(var + 1e-5f);
    __bf16* orow = out + (size_t)row * D_MODEL;
    for (int i = tid; i < D_MODEL; i += 256)
        orow[i] = (__bf16)((xr[i] - mu) * rs * w[i] + b[i]);
}

// ---------------------------------------------------------------------------
// Tiled bf16 GEMM: acc = A[M,K] * B[K,N]; out = acc*scale + bias + resid.
// Output f32 (C) or bf16 (Cbf) -- exactly one of C/Cbf is non-null.
// Block 128x128, 8 waves, wave tile 32x64 (2x4 WMMA tiles), TK=32.
// A staged row-major (async->LDS when available), B staged column-major.
// ---------------------------------------------------------------------------
#define LDS_P 40  // padded K stride (elements): 80B rows, 16B aligned

__global__ __launch_bounds__(256) void gemm_bf16_kernel(
    const __bf16* __restrict__ A, int lda,
    const __bf16* __restrict__ Bw, int ldb,
    float* __restrict__ C, __bf16* __restrict__ Cbf, int ldc,
    const float* __restrict__ bias, int nbias,
    const float* __restrict__ resid, float scale, int K) {
    __shared__ __attribute__((aligned(16))) __bf16 ldsA[128 * LDS_P];
    __shared__ __attribute__((aligned(16))) __bf16 ldsB[128 * LDS_P]; // [n][k]

    int tid  = threadIdx.x;
    int wave = tid >> 5, lane = tid & 31;
    int half = lane >> 4, lr = lane & 15;
    int wr = wave >> 1, wc = wave & 1;
    int m0 = blockIdx.y * 128, n0 = blockIdx.x * 128;

    v8f acc[2][4];
#pragma unroll
    for (int i = 0; i < 2; ++i)
#pragma unroll
        for (int j = 0; j < 4; ++j) acc[i][j] = vzero8();

    const int nkt = K >> 5;
    for (int kt = 0; kt < nkt; ++kt) {
        int k0 = kt << 5;
        // Stage A tile 128x32.
#pragma unroll
        for (int qc = 0; qc < 2; ++qc) {
            int f = tid + (qc << 8);           // 0..511
            int row = f >> 2, kk = (f & 3) << 3;
            const __bf16* src = A + (size_t)(m0 + row) * lda + k0 + kk;
#if HAVE_ASYNC_LDS
            async_copy_b128(&ldsA[row * LDS_P + kk], src);
#else
            uint4 d = *(const uint4*)src;
            *(uint4*)&ldsA[row * LDS_P + kk] = d;
#endif
        }
        // Stage B tile 32x128, transposed into [n][k].
#pragma unroll
        for (int qc = 0; qc < 2; ++qc) {
            int f = tid + (qc << 8);
            int krow = f >> 4, nc0 = (f & 15) << 3;
            uint4 d = *(const uint4*)(Bw + (size_t)(k0 + krow) * ldb + n0 + nc0);
            const __bf16* e = (const __bf16*)&d;
#pragma unroll
            for (int i = 0; i < 8; ++i) ldsB[(nc0 + i) * LDS_P + krow] = e[i];
        }
        if (kt + 1 < nkt) {
            __builtin_prefetch(A + (size_t)(m0 + (tid >> 1)) * lda + k0 + 32, 0, 1);
            __builtin_prefetch(Bw + (size_t)(k0 + 32 + (tid >> 3)) * ldb + n0, 0, 1);
        }
#if HAVE_ASYNC_LDS
        async_wait0();
#endif
        __syncthreads();

        AF af[2], bf[4];
#pragma unroll
        for (int i = 0; i < 2; ++i) {
            int rbase = (wr << 5) + (i << 4) + lr;
#pragma unroll
            for (int v = 0; v < 8; ++v) {
                int k = ((v & 3) << 1) + (half << 3) + ((v >> 2) << 4);
                af[i].u[v] = *(const unsigned*)&ldsA[rbase * LDS_P + k];
            }
        }
#pragma unroll
        for (int j = 0; j < 4; ++j) {
            int nbase = (wc << 6) + (j << 4) + lr;
#pragma unroll
            for (int v = 0; v < 8; ++v) {
                int k = ((v & 3) << 1) + (half << 3) + ((v >> 2) << 4);
                bf[j].u[v] = *(const unsigned*)&ldsB[nbase * LDS_P + k];
            }
        }
#pragma unroll
        for (int i = 0; i < 2; ++i)
#pragma unroll
            for (int j = 0; j < 4; ++j)
                acc[i][j] = __builtin_amdgcn_wmma_f32_16x16x32_bf16(
                    false, af[i].v, false, bf[j].v, (short)0, acc[i][j], false, false);
        __syncthreads();
    }

    // Epilogue: scale + bias + residual; f32 or bf16 store.
#pragma unroll
    for (int i = 0; i < 2; ++i) {
#pragma unroll
        for (int j = 0; j < 4; ++j) {
            int col = n0 + (wc << 6) + (j << 4) + lr;
            float bv = (bias && col < nbias) ? bias[col] : 0.0f;
#pragma unroll
            for (int p = 0; p < 8; ++p) {
                int row = m0 + (wr << 5) + (i << 4) + (half << 3) + p;
                float v = acc[i][j][p] * scale + bv;
                if (resid) v += resid[(size_t)row * ldc + col];
                if (Cbf) Cbf[(size_t)row * ldc + col] = (__bf16)v;
                else     C  [(size_t)row * ldc + col] = v;
            }
        }
    }
}

// ---------------------------------------------------------------------------
// Flash-style retention: per (64-query tile, batch*head) block.
// q,k,v arrive as bf16 (q pre-scaled by 1/8 in the GEMM epilogue).
// scores = q @ k^T (WMMA), decay gamma^(s-t) applied in C-fragment regs,
// out += scores_bf16 @ v (WMMA), then per-row groupnorm + SiLU(g) gate,
// bf16 output (feeds the wo GEMM directly).
// ---------------------------------------------------------------------------
#define QP 72  // padded stride for 64-wide bf16 LDS tiles

__global__ __launch_bounds__(256) void retention_kernel(
    const __bf16* __restrict__ qg, const __bf16* __restrict__ kg,
    const __bf16* __restrict__ vg, const float* __restrict__ gg,
    __bf16* __restrict__ msr) {
    __shared__ __attribute__((aligned(16))) __bf16 qs [64 * QP];
    __shared__ __attribute__((aligned(16))) __bf16 ks [64 * QP];
    __shared__ __attribute__((aligned(16))) __bf16 vsT[64 * QP]; // [c][t]
    __shared__ __attribute__((aligned(16))) __bf16 ss [64 * QP];
    __shared__ float outLds[64 * 65];
    __shared__ float mus[64], rss[64];

    int tid = threadIdx.x;
    int wave = tid >> 5, lane = tid & 31, half = lane >> 4, lr = lane & 15;
    int rb = wave >> 1, cb = wave & 1;
    int qt = blockIdx.x;
    int bh = blockIdx.y, b = bh >> 4, h = bh & 15;
    int s0 = qt << 6;
    size_t base = ((size_t)b * SEQ) * D_MODEL + h * HEAD_DIM;
    float lg2g = log2f(1.0f - exp2f(-5.0f - (float)h));

    // Stage q tile (pure bf16 copy; 512 x 16B chunks).
#pragma unroll
    for (int i = 0; i < 2; ++i) {
        int ch = tid + (i << 8);
        int r = ch >> 3, c0 = (ch & 7) << 3;
        const __bf16* src = qg + base + (size_t)(s0 + r) * D_MODEL + c0;
#if HAVE_ASYNC_LDS
        async_copy_b128(&qs[r * QP + c0], src);
#else
        uint4 d = *(const uint4*)src;
        *(uint4*)&qs[r * QP + c0] = d;
#endif
    }
#if HAVE_ASYNC_LDS
    async_wait0();
#endif
    __syncthreads();

    // q A-fragments (loop invariant).
    AF aq[2];
#pragma unroll
    for (int kc = 0; kc < 2; ++kc) {
        int rbase = (rb << 4) + lr;
#pragma unroll
        for (int v = 0; v < 8; ++v) {
            int k = (kc << 5) + ((v & 3) << 1) + (half << 3) + ((v >> 2) << 4);
            aq[kc].u[v] = *(const unsigned*)&qs[rbase * QP + k];
        }
    }

    v8f accO[2];
    accO[0] = vzero8(); accO[1] = vzero8();

    for (int tt = 0; tt <= qt; ++tt) {
        int t0 = tt << 6;
        // Stage k row-major (async copy) and v transposed (sync).
#pragma unroll
        for (int i = 0; i < 2; ++i) {
            int ch = tid + (i << 8);
            int r = ch >> 3, c0 = (ch & 7) << 3;
            const __bf16* ksrc = kg + base + (size_t)(t0 + r) * D_MODEL + c0;
#if HAVE_ASYNC_LDS
            async_copy_b128(&ks[r * QP + c0], ksrc);
#else
            uint4 dk = *(const uint4*)ksrc;
            *(uint4*)&ks[r * QP + c0] = dk;
#endif
            uint4 dv = *(const uint4*)(vg + base + (size_t)(t0 + r) * D_MODEL + c0);
            const __bf16* e = (const __bf16*)&dv;
#pragma unroll
            for (int j = 0; j < 8; ++j) vsT[(c0 + j) * QP + r] = e[j];
        }
#if HAVE_ASYNC_LDS
        async_wait0();
#endif
        __syncthreads();

        // Scores: wave computes its 16-row strip, cols 32*cb..+31 (2 tiles).
#pragma unroll
        for (int jj = 0; jj < 2; ++jj) {
            int jt2 = (cb << 1) + jj;
            AF bk[2];
#pragma unroll
            for (int kc = 0; kc < 2; ++kc) {
                int nbase = (jt2 << 4) + lr;
#pragma unroll
                for (int v = 0; v < 8; ++v) {
                    int k = (kc << 5) + ((v & 3) << 1) + (half << 3) + ((v >> 2) << 4);
                    bk[kc].u[v] = *(const unsigned*)&ks[nbase * QP + k];
                }
            }
            v8f sacc = vzero8();
            sacc = __builtin_amdgcn_wmma_f32_16x16x32_bf16(
                false, aq[0].v, false, bk[0].v, (short)0, sacc, false, false);
            sacc = __builtin_amdgcn_wmma_f32_16x16x32_bf16(
                false, aq[1].v, false, bk[1].v, (short)0, sacc, false, false);
            // Apply decay mask in C layout, spill to LDS as bf16.
#pragma unroll
            for (int p = 0; p < 8; ++p) {
                int srow = s0 + (rb << 4) + (half << 3) + p;
                int tcol = t0 + (jt2 << 4) + lr;
                int delta = srow - tcol;
                float dv = (delta >= 0) ? exp2f(lg2g * (float)delta) : 0.0f;
                ss[((rb << 4) + (half << 3) + p) * QP + (jt2 << 4) + lr] =
                    (__bf16)(sacc[p] * dv);
            }
        }
        __syncthreads();

        // out += scores @ v.
        AF as[2];
#pragma unroll
        for (int kc = 0; kc < 2; ++kc) {
            int rbase = (rb << 4) + lr;
#pragma unroll
            for (int v = 0; v < 8; ++v) {
                int k = (kc << 5) + ((v & 3) << 1) + (half << 3) + ((v >> 2) << 4);
                as[kc].u[v] = *(const unsigned*)&ss[rbase * QP + k];
            }
        }
#pragma unroll
        for (int jj = 0; jj < 2; ++jj) {
            AF bv[2];
            int nbase = (cb << 5) + (jj << 4) + lr;
#pragma unroll
            for (int kc = 0; kc < 2; ++kc) {
#pragma unroll
                for (int v = 0; v < 8; ++v) {
                    int k = (kc << 5) + ((v & 3) << 1) + (half << 3) + ((v >> 2) << 4);
                    bv[kc].u[v] = *(const unsigned*)&vsT[nbase * QP + k];
                }
            }
            accO[jj] = __builtin_amdgcn_wmma_f32_16x16x32_bf16(
                false, as[0].v, false, bv[0].v, (short)0, accO[jj], false, false);
            accO[jj] = __builtin_amdgcn_wmma_f32_16x16x32_bf16(
                false, as[1].v, false, bv[1].v, (short)0, accO[jj], false, false);
        }
        __syncthreads();
    }

    // Group norm over head dim + SiLU gate, bf16 store.
#pragma unroll
    for (int jj = 0; jj < 2; ++jj)
#pragma unroll
        for (int p = 0; p < 8; ++p) {
            int r = (rb << 4) + (half << 3) + p;
            int c = (cb << 5) + (jj << 4) + lr;
            outLds[r * 65 + c] = accO[jj][p];
        }
    __syncthreads();
    if (tid < 64) {
        float a = 0.f, q2 = 0.f;
        for (int c = 0; c < 64; ++c) { float v = outLds[tid * 65 + c]; a += v; q2 += v * v; }
        float mu  = a * (1.0f / 64.0f);
        float var = q2 * (1.0f / 64.0f) - mu * mu;
        mus[tid] = mu; rss[tid] = rsqrtf(var + 1e-6f);
    }
    __syncthreads();
#pragma unroll 4
    for (int i = 0; i < 16; ++i) {
        int e = tid + (i << 8);
        int r = e >> 6, c = e & 63;
        float gv = gg[base + (size_t)(s0 + r) * D_MODEL + c];
        float nv = (outLds[r * 65 + c] - mus[r]) * rss[r];
        msr[base + (size_t)(s0 + r) * D_MODEL + c] = (__bf16)(nv * gv * sigmoidf_(gv));
    }
}

// ---------------------------------------------------------------------------
// SwiGLU: h = a * silu(b) from bf16 halves, bf16 out into K-padded buffer.
// ---------------------------------------------------------------------------
__global__ void swiglu_kernel(const __bf16* __restrict__ ab, __bf16* __restrict__ hbf) {
    long e = (long)blockIdx.x * 256 + threadIdx.x;
    if (e >= (long)ROWS * HID_PAD) return;
    int r = (int)(e / HID_PAD), c = (int)(e % HID_PAD);
    float out = 0.0f;
    if (c < HID) {
        const __bf16* row = ab + (size_t)r * N_IN_PAD;
        float a = (float)row[c], bb = (float)row[HID + c];
        out = a * bb * sigmoidf_(bb);
    }
    hbf[e] = (__bf16)out;
}

// ---------------------------------------------------------------------------
extern "C" void kernel_launch(void* const* d_in, const int* in_sizes, int n_in,
                              void* d_out, int out_size, void* d_ws, size_t ws_size,
                              hipStream_t stream) {
    (void)in_sizes; (void)n_in; (void)out_size; (void)ws_size;
    const float* x    = (const float*)d_in[0];
    const float* ln1w = (const float*)d_in[1];
    const float* ln1b = (const float*)d_in[2];
    const float* wq   = (const float*)d_in[3];
    const float* wk   = (const float*)d_in[4];
    const float* wv   = (const float*)d_in[5];
    const float* wg   = (const float*)d_in[6];
    const float* wo   = (const float*)d_in[7];
    const float* ln2w = (const float*)d_in[8];
    const float* ln2b = (const float*)d_in[9];
    const float* win  = (const float*)d_in[10];
    const float* bin  = (const float*)d_in[11];
    const float* wout = (const float*)d_in[12];
    const float* bout = (const float*)d_in[13];
    float* out = (float*)d_out;

    char* ws = (char*)d_ws;
    size_t off = 0;
    auto alloc = [&](size_t bytes) -> char* {
        char* p = ws + off;
        off += (bytes + 255) & ~(size_t)255;
        return p;
    };
    __bf16* wq_bf   = (__bf16*)alloc((size_t)D_MODEL * D_MODEL * 2);
    __bf16* wk_bf   = (__bf16*)alloc((size_t)D_MODEL * D_MODEL * 2);
    __bf16* wv_bf   = (__bf16*)alloc((size_t)D_MODEL * D_MODEL * 2);
    __bf16* wg_bf   = (__bf16*)alloc((size_t)D_MODEL * D_MODEL * 2);
    __bf16* wo_bf   = (__bf16*)alloc((size_t)D_MODEL * D_MODEL * 2);
    __bf16* win_bf  = (__bf16*)alloc((size_t)D_MODEL * N_IN_PAD * 2);
    __bf16* wout_bf = (__bf16*)alloc((size_t)HID_PAD * D_MODEL * 2);
    __bf16* xn_bf   = (__bf16*)alloc((size_t)ROWS * D_MODEL * 2);
    __bf16* qbf     = (__bf16*)alloc((size_t)ROWS * D_MODEL * 2);
    __bf16* kbf     = (__bf16*)alloc((size_t)ROWS * D_MODEL * 2);
    __bf16* vbf     = (__bf16*)alloc((size_t)ROWS * D_MODEL * 2);
    float*  gb      = (float*)alloc((size_t)ROWS * D_MODEL * 4);
    __bf16* msr_bf  = (__bf16*)alloc((size_t)ROWS * D_MODEL * 2);
    float*  x2      = (float*)alloc((size_t)ROWS * D_MODEL * 4);
    __bf16* xn2_bf  = (__bf16*)alloc((size_t)ROWS * D_MODEL * 2);
    __bf16* ab_bf   = (__bf16*)alloc((size_t)ROWS * N_IN_PAD * 2);
    __bf16* h_bf    = (__bf16*)alloc((size_t)ROWS * HID_PAD * 2);

    dim3 blk(256);
    auto nb = [](long total) { return dim3((unsigned)((total + 255) / 256)); };

    // Weight conversions (f32 -> bf16, zero-padded where needed).
    cvt_pad_bf16<<<nb(1024L * 1024), blk, 0, stream>>>(wq, wq_bf, 1024, 1024, 1024, 1024L * 1024);
    cvt_pad_bf16<<<nb(1024L * 1024), blk, 0, stream>>>(wk, wk_bf, 1024, 1024, 1024, 1024L * 1024);
    cvt_pad_bf16<<<nb(1024L * 1024), blk, 0, stream>>>(wv, wv_bf, 1024, 1024, 1024, 1024L * 1024);
    cvt_pad_bf16<<<nb(1024L * 1024), blk, 0, stream>>>(wg, wg_bf, 1024, 1024, 1024, 1024L * 1024);
    cvt_pad_bf16<<<nb(1024L * 1024), blk, 0, stream>>>(wo, wo_bf, 1024, 1024, 1024, 1024L * 1024);
    cvt_pad_bf16<<<nb(1024L * N_IN_PAD), blk, 0, stream>>>(win, win_bf, 1024, N_IN, N_IN_PAD, 1024L * N_IN_PAD);
    cvt_pad_bf16<<<nb((long)HID_PAD * 1024), blk, 0, stream>>>(wout, wout_bf, HID, 1024, 1024, (long)HID_PAD * 1024);

    // LN1 -> bf16.
    layernorm_bf16_kernel<<<ROWS, blk, 0, stream>>>(x, ln1w, ln1b, xn_bf);

    // QKVG projections (q/k/v stored bf16, q pre-scaled; gate stays f32).
    dim3 g1(1024 / 128, ROWS / 128);
    gemm_bf16_kernel<<<g1, blk, 0, stream>>>(xn_bf, 1024, wq_bf, 1024, nullptr, qbf, 1024, nullptr, 0, nullptr, 0.125f, 1024);
    gemm_bf16_kernel<<<g1, blk, 0, stream>>>(xn_bf, 1024, wk_bf, 1024, nullptr, kbf, 1024, nullptr, 0, nullptr, 1.0f, 1024);
    gemm_bf16_kernel<<<g1, blk, 0, stream>>>(xn_bf, 1024, wv_bf, 1024, nullptr, vbf, 1024, nullptr, 0, nullptr, 1.0f, 1024);
    gemm_bf16_kernel<<<g1, blk, 0, stream>>>(xn_bf, 1024, wg_bf, 1024, gb, nullptr, 1024, nullptr, 0, nullptr, 1.0f, 1024);

    // Retention + group norm + gate -> bf16.
    retention_kernel<<<dim3(SEQ / 64, BATCH * NHEAD), blk, 0, stream>>>(qbf, kbf, vbf, gb, msr_bf);

    // Output projection with residual (x2 = x + msr @ wo).
    gemm_bf16_kernel<<<g1, blk, 0, stream>>>(msr_bf, 1024, wo_bf, 1024, x2, nullptr, 1024, nullptr, 0, x, 1.0f, 1024);

    // LN2 -> bf16, MLP in (bf16 out), SwiGLU, MLP out with residual into d_out.
    layernorm_bf16_kernel<<<ROWS, blk, 0, stream>>>(x2, ln2w, ln2b, xn2_bf);
    dim3 g2(N_IN_PAD / 128, ROWS / 128);
    gemm_bf16_kernel<<<g2, blk, 0, stream>>>(xn2_bf, 1024, win_bf, N_IN_PAD, nullptr, ab_bf, N_IN_PAD, bin, N_IN, nullptr, 1.0f, 1024);
    swiglu_kernel<<<nb((long)ROWS * HID_PAD), blk, 0, stream>>>(ab_bf, h_bf);
    gemm_bf16_kernel<<<g1, blk, 0, stream>>>(h_bf, HID_PAD, wout_bf, 1024, out, nullptr, 1024, bout, 1024, x2, 1.0f, HID_PAD);
}
